// ViTBeans_74569222193922
// MI455X (gfx1250) — matrix operand order repaired and checked by reading, not compiled, not run on hardware
//
#include <hip/hip_runtime.h>
#include <math.h>

#define NE 16
#define NH 8
#define DD 1024
#define EE 64
#define SSZ 128
#define BB 8
#define PP 576
#define M_TOTAL (BB * PP) /* 4608 */

typedef __attribute__((ext_vector_type(16))) _Float16 v16h;
typedef __attribute__((ext_vector_type(8)))  _Float16 v8h;
typedef __attribute__((ext_vector_type(8)))  float    v8f;
typedef __attribute__((ext_vector_type(4)))  unsigned int u32x4;
typedef __attribute__((ext_vector_type(4)))  int i32x4;
typedef __attribute__((ext_vector_type(8)))  int i32x8;

#if defined(__has_builtin)
#if __has_builtin(__builtin_amdgcn_tensor_load_to_lds)
#define HAVE_TDM 1
#else
#define HAVE_TDM 0
#endif
#else
#define HAVE_TDM 0
#endif

__device__ __forceinline__ v8f wmma_f16(v16h a, v16h b, v8f c) {
  return __builtin_amdgcn_wmma_f32_16x16x32_f16(false, a, false, b, (short)0, c,
                                                false, false);
}

// A fragment (16x32 f16) from f32 rows (with row scale): lane holds row lane&15;
// K set {0..7,16..23} (lanes 0-15) / {8..15,24..31} (lanes 16-31), ISA 7.12.2.
__device__ __forceinline__ v16h a_frag(const float* rowptr, int kb, int lane,
                                       float scl) {
  const int base = (lane & 16) ? 8 : 0;
  v16h a;
#pragma unroll
  for (int i = 0; i < 16; ++i) {
    int k = kb + base + i + ((i >= 8) ? 8 : 0);
    a[i] = (_Float16)(rowptr[k] * scl);
  }
  return a;
}

// A fragment from f16 rows: two aligned 16B vector loads.
__device__ __forceinline__ v16h a_frag_h(const _Float16* rowptr, int kb,
                                         int lane) {
  const int base = kb + ((lane & 16) ? 8 : 0);
  v8h lo = *(const v8h*)(rowptr + base);
  v8h hi = *(const v8h*)(rowptr + base + 16);
  v16h r;
#pragma unroll
  for (int i = 0; i < 8; ++i) {
    r[i] = lo[i];
    r[i + 8] = hi[i];
  }
  return r;
}

// B fragment (32x16 f16) where this lane's column is contiguous in K
// (source stored N-major, K contiguous): two aligned 16B vector loads,
// K set {0..7,16..23} (lanes 0-15) / {8..15,24..31} (lanes 16-31).
__device__ __forceinline__ v16h b_frag_h(const _Float16* rowptr, int kb,
                                         int lane) {
  const int kb2 = kb + ((lane & 16) ? 8 : 0);
  v8h lo = *(const v8h*)(rowptr + kb2);
  v8h hi = *(const v8h*)(rowptr + kb2 + 16);
  v16h r;
#pragma unroll
  for (int i = 0; i < 8; ++i) {
    r[i] = lo[i];
    r[i + 8] = hi[i];
  }
  return r;
}

#if HAVE_TDM
// 2D f16 tile DMA (TDM): rows x cols elements, row pitch stride_elems.
__device__ __forceinline__ void tdm_load_2d_f16(unsigned lds_off,
                                                const void* gptr, unsigned rows,
                                                unsigned cols,
                                                unsigned long long stride_e) {
  unsigned long long ga = (unsigned long long)(size_t)gptr;
  u32x4 g0;
  g0[0] = 1u;                                  // count=1, user descriptor
  g0[1] = lds_off;                             // D#.lds_addr (bytes)
  g0[2] = (unsigned)ga;                        // global_addr[31:0]
  g0[3] = (unsigned)((ga >> 32) & 0x01FFFFFFu) | (2u << 30);  // [56:32]|type=2
  i32x8 g1;
  g1[0] = 0x00010000;                                    // data_size=1 (2B)
  g1[1] = (int)((cols & 0xFFFFu) << 16);                 // tensor_dim0 lo
  g1[2] = (int)(((cols >> 16) & 0xFFFFu) | ((rows & 0xFFFFu) << 16));
  g1[3] = (int)(((rows >> 16) & 0xFFFFu) | ((cols & 0xFFFFu) << 16)); // tile0
  g1[4] = (int)(rows & 0xFFFFu);                         // tile_dim1
  g1[5] = (int)(unsigned)(stride_e & 0xFFFFFFFFull);     // dim0_stride lo32
  g1[6] = (int)(unsigned)((stride_e >> 32) & 0xFFFFull); // dim0_stride hi16
  g1[7] = 0;
  i32x4 z4 = {0, 0, 0, 0};
#if __clang_major__ >= 23
  i32x8 z8 = {0, 0, 0, 0, 0, 0, 0, 0};
  __builtin_amdgcn_tensor_load_to_lds(g0, g1, z4, z4, z8, 0);
#else
  __builtin_amdgcn_tensor_load_to_lds(g0, g1, z4, z4, 0);
#endif
}
#endif

// ---------------------------------------------------------------- masks/votes
__global__ void mask_kernel(const float* __restrict__ fp, float* maskf,
                            float* votes) {
  int p = threadIdx.x;
  if (p >= PP) return;
  float f = fp[p];
  float v = 0.f;
  for (int e = 0; e < NE; ++e) {
    float mn = fmaxf(0.f, e * 0.0625f - 0.015625f);
    float mx = fminf(1.f, (e + 1) * 0.0625f + 0.015625f);
    bool in = (e == NE - 1) ? (f >= mn && f <= mx) : (f >= mn && f < mx);
    float mf = in ? 1.f : 0.f;
    maskf[e * PP + p] = mf;
    v += mf;
  }
  votes[p] = v;
}

// --------------------------- tokens f32 -> f16 (converted once, reused 6x)
__global__ void convtok_kernel(const float* __restrict__ tokens,
                               _Float16* tokh) {
  size_t i = (size_t)blockIdx.x * blockDim.x + threadIdx.x;
  tokh[i] = (_Float16)tokens[i];
}

// -------------------- weight pre-conversion to f16, K-contiguous (transposed)
__global__ void convw_kernel(const float* __restrict__ qW,
                             const float* __restrict__ kW,
                             const float* __restrict__ vW,
                             const float* __restrict__ oW, _Float16* qWT,
                             _Float16* kWT, _Float16* vWT, _Float16* oWT) {
  const int e = blockIdx.y;
  const int id = blockIdx.x * blockDim.x + threadIdx.x;  // 8192 = 128*64
  {
    int k = id / EE, n = id % EE;  // qW/kW/vW: [e][k=128][n=64]
    size_t src = (size_t)e * SSZ * EE + (size_t)k * EE + n;
    size_t dst = (size_t)e * EE * SSZ + (size_t)n * SSZ + k;  // [e][n][k]
    qWT[dst] = (_Float16)qW[src];
    kWT[dst] = (_Float16)kW[src];
    vWT[dst] = (_Float16)vW[src];
  }
  {
    int k = id / SSZ, n = id % SSZ;  // oW: [e][k=64][n=128]
    oWT[(size_t)e * SSZ * EE + (size_t)n * EE + k] =
        (_Float16)oW[(size_t)e * EE * SSZ + (size_t)k * SSZ + n];
  }
}

// ------------------------------------------------- Q/K/V GEMMs (WMMA, 1 wave)
__global__ void qkv_kernel(const _Float16* __restrict__ tokh,
                           const _Float16* __restrict__ qWT,
                           const _Float16* __restrict__ kWT,
                           const _Float16* __restrict__ vWT,
                           const float* __restrict__ qb,
                           const float* __restrict__ kb_,
                           const float* __restrict__ vb, float* Qb, float* Kb,
                           float* Vb, _Float16* Kh, _Float16* VhT) {
  const int lane = threadIdx.x;
  const int mt = blockIdx.x, e = blockIdx.y, z = blockIdx.z;
  const int head = e & 7;
  const int m = mt * 16 + (lane & 15);
  const int b = m / PP, p = m % PP;
  const _Float16* arow = tokh + ((size_t)b * PP + p) * DD + head * SSZ;
  const _Float16* W = (z == 0 ? qWT : z == 1 ? kWT : vWT) + (size_t)e * EE * SSZ;
  const float* bias = (z == 0 ? qb : z == 1 ? kb_ : vb) + (size_t)e * EE;
  float* Out = (z == 0 ? Qb : z == 1 ? Kb : Vb);

  const v8f vz = {};
  v8f acc[4];
#pragma unroll
  for (int nt = 0; nt < 4; ++nt) acc[nt] = vz;

#pragma unroll
  for (int ks = 0; ks < 4; ++ks) {
    v16h a = a_frag_h(arow, ks * 32, lane);
#pragma unroll
    for (int nt = 0; nt < 4; ++nt) {
      const _Float16* wrow = W + (size_t)(nt * 16 + (lane & 15)) * SSZ;
      acc[nt] = wmma_f16(a, b_frag_h(wrow, ks * 32, lane), acc[nt]);
    }
  }
#pragma unroll
  for (int j = 0; j < 8; ++j) {
    int row = mt * 16 + j + ((lane >> 4) << 3);
#pragma unroll
    for (int nt = 0; nt < 4; ++nt) {
      int colg = nt * 16 + (lane & 15);
      float v = acc[nt][j] + bias[colg];
      Out[((size_t)e * M_TOTAL + row) * EE + colg] = v;
      if (z == 1)
        Kh[((size_t)e * M_TOTAL + row) * EE + colg] = (_Float16)v;
      else if (z == 2)
        VhT[((size_t)e * EE + colg) * M_TOTAL + row] = (_Float16)v;
    }
  }
}

// ----------------------------------------- mini-MHA precompute (per expert)
__global__ void mha_pre_kernel(const float* __restrict__ penta,
                               const float* __restrict__ mha_w,
                               const float* __restrict__ mha_b,
                               const float* __restrict__ mha_ow, float* kq,
                               float* vo, float* sb) {
  const int e = blockIdx.x, t = threadIdx.x;  // 64 threads
  __shared__ float kk[5][EE];
  __shared__ float vv[5][EE];
  const float* pe = penta + (size_t)e * 5 * EE;
  const float* W = mha_w + (size_t)e * 3 * EE * EE;
  const float* bbp = mha_b + (size_t)e * 3 * EE;
#pragma unroll
  for (int v = 0; v < 5; ++v) {
    float accK = 0.f, accV = 0.f;
    for (int i = 0; i < EE; ++i) {
      float pv = pe[v * EE + i];
      accK += pv * W[(EE + t) * EE + i];
      accV += pv * W[(2 * EE + t) * EE + i];
    }
    kk[v][t] = accK + bbp[EE + t];
    vv[v][t] = accV + bbp[2 * EE + t];
  }
  __syncthreads();
  const float* OW = mha_ow + (size_t)e * EE * EE;
#pragma unroll
  for (int v = 0; v < 5; ++v) {
    float aq = 0.f, ao = 0.f;
    for (int j = 0; j < EE; ++j) {
      aq += kk[v][j] * W[j * EE + t];   // Wq^T kk
      ao += vv[v][j] * OW[t * EE + j];  // ow row t
    }
    kq[((size_t)e * 5 + v) * EE + t] = aq;
    vo[((size_t)e * 5 + v) * EE + t] = ao;
  }
  if (t < 5) {
    float s = 0.f;
    for (int j = 0; j < EE; ++j) s += bbp[j] * kk[t][j];
    sb[e * 5 + t] = s;
  }
}

__global__ void mha_apply_kernel(float* Qb, _Float16* Qh,
                                 const float* __restrict__ kq,
                                 const float* __restrict__ vo,
                                 const float* __restrict__ sb,
                                 const float* __restrict__ mha_ob) {
  int id = blockIdx.x * blockDim.x + threadIdx.x;  // NE*M_TOTAL
  int e = id / M_TOTAL, m = id % M_TOTAL;
  float* Qrow = Qb + ((size_t)e * M_TOTAL + m) * EE;
  _Float16* Qhrow = Qh + ((size_t)e * M_TOTAL + m) * EE;
  float s[5];
#pragma unroll
  for (int v = 0; v < 5; ++v) {
    float acc = sb[e * 5 + v];
    const float* kqv = kq + ((size_t)e * 5 + v) * EE;
    for (int k = 0; k < EE; ++k) acc += Qrow[k] * kqv[k];
    s[v] = acc * 0.125f;
  }
  float mx = s[0];
#pragma unroll
  for (int v = 1; v < 5; ++v) mx = fmaxf(mx, s[v]);
  float sum = 0.f;
#pragma unroll
  for (int v = 0; v < 5; ++v) {
    s[v] = __expf(s[v] - mx);
    sum += s[v];
  }
  float inv = 1.f / sum;
  const float* obp = mha_ob + (size_t)e * EE;
  for (int k = 0; k < EE; ++k) {
    float d = 0.f;
#pragma unroll
    for (int v = 0; v < 5; ++v) d += s[v] * inv * vo[((size_t)e * 5 + v) * EE + k];
    float val = Qrow[k] + d + obp[k];
    Qrow[k] = val;
    Qhrow[k] = (_Float16)val;
  }
}

// -------------------------------------------------- Kg / Vg gate scalars
__global__ void kgvg_kernel(const float* __restrict__ penta,
                            const float* __restrict__ Kb,
                            const float* __restrict__ Vb, float* Kg, float* Vg) {
  const int e = blockIdx.y, t = threadIdx.x;
  __shared__ float r[EE];
  __shared__ float pn[5];
  const float* pe = penta + (size_t)e * 5 * EE;
  if (t < 5) {
    float s = 0.f;
    for (int i = 0; i < EE; ++i) s += pe[t * EE + i] * pe[t * EE + i];
    pn[t] = fmaxf(sqrtf(s), 1e-12f);
  }
  __syncthreads();
  if (t < EE) {
    const float rw[5] = {1.0f, -0.75f, 0.75f, 0.75f, -0.75f};
    float s = 0.f;
#pragma unroll
    for (int v = 0; v < 5; ++v) s += rw[v] * pe[v * EE + t] / pn[v];
    r[t] = s;
  }
  __syncthreads();
  const int m = blockIdx.x * blockDim.x + t;
  const float* krow = Kb + ((size_t)e * M_TOTAL + m) * EE;
  const float* vrow = Vb + ((size_t)e * M_TOTAL + m) * EE;
  float nk = 0.f, nv = 0.f, dk = 0.f, dv = 0.f;
  for (int i = 0; i < EE; ++i) {
    float kv = krow[i], vx = vrow[i];
    nk += kv * kv;
    nv += vx * vx;
    dk += kv * r[i];
    dv += vx * r[i];
  }
  Kg[(size_t)e * M_TOTAL + m] = dk / fmaxf(sqrtf(nk), 1e-12f);
  Vg[(size_t)e * M_TOTAL + m] = dv / fmaxf(sqrtf(nv), 1e-12f);
}

// ------- fused attention: TDM-staged K/V^T in LDS, QK^T WMMA -> softmax -> PV
// 4 waves / block, each wave owns one 16-query tile; K (576x64 f16) and
// V^T (64x576 f16) staged once per block via Tensor Data Mover.
__global__ void attn_kernel(const _Float16* __restrict__ Qh,
                            const _Float16* __restrict__ Kh,
                            const _Float16* __restrict__ VhT,
                            const float* __restrict__ Kg,
                            const float* __restrict__ Vg,
                            const float* __restrict__ maskf, float* Ob) {
  extern __shared__ char smem[];
  _Float16* KL = (_Float16*)smem;                 // 576*64 f16 = 72KB
  _Float16* VL = (_Float16*)(smem + PP * EE * 2); // 64*576 f16 = 72KB
  float* Sall = (float*)(smem + 2 * PP * EE * 2); // 4 * 16*576 f32 = 144KB
  const int tid = threadIdx.x;
  const int w = tid >> 5, lane = tid & 31;
  const int b = blockIdx.y, e = blockIdx.z;
  const int mt = blockIdx.x * 4 + w;
  float* S = Sall + (size_t)w * 16 * PP;
  const size_t eb = (size_t)e * M_TOTAL + (size_t)b * PP;

#if HAVE_TDM
  if (w == 0) {
    tdm_load_2d_f16((unsigned)(size_t)KL, Kh + eb * EE, PP, EE, EE);
    tdm_load_2d_f16((unsigned)(size_t)VL,
                    VhT + (size_t)e * EE * M_TOTAL + (size_t)b * PP, EE, PP,
                    M_TOTAL);
    __builtin_amdgcn_s_wait_tensorcnt(0);
  }
#else
  for (int i = tid; i < PP * EE; i += 128) KL[i] = Kh[eb * EE + i];
  for (int i = tid; i < EE * PP; i += 128) {
    int f = i / PP, c = i % PP;
    VL[i] = VhT[((size_t)e * EE + f) * M_TOTAL + (size_t)b * PP + c];
  }
#endif
  __syncthreads();

  const _Float16* arow = Qh + (eb + (size_t)(mt * 16 + (lane & 15))) * EE;
  const v16h a0 = a_frag_h(arow, 0, lane);
  const v16h a1 = a_frag_h(arow, 32, lane);
  const float scale = 0.125f;

  // pass 1: scores into this wave's LDS strip
  for (int nt = 0; nt < PP / 16; ++nt) {
    const _Float16* krow = KL + (size_t)(nt * 16 + (lane & 15)) * EE;
    v8f acc = {};
    acc = wmma_f16(a0, b_frag_h(krow, 0, lane), acc);
    acc = wmma_f16(a1, b_frag_h(krow, 32, lane), acc);
    const int col = lane & 15;
    const int key = nt * 16 + col;
    const float kg = Kg[eb + key];
    const float mk = maskf[e * PP + key];
#pragma unroll
    for (int j = 0; j < 8; ++j) {
      int row = j + ((lane >> 4) << 3);
      float v = acc[j] * scale * kg;
      S[row * PP + key] = (mk > 0.f) ? v : -1e9f;
    }
  }

  // softmax per query row over 576 keys (wave cooperative, own strip)
  for (int r = 0; r < 16; ++r) {
    float mx = -3.4e38f;
    for (int c = lane; c < PP; c += 32) mx = fmaxf(mx, S[r * PP + c]);
#pragma unroll
    for (int off = 16; off > 0; off >>= 1) mx = fmaxf(mx, __shfl_xor(mx, off, 32));
    float sum = 0.f;
    for (int c = lane; c < PP; c += 32) {
      float ev = __expf(S[r * PP + c] - mx);
      S[r * PP + c] = ev;
      sum += ev;
    }
#pragma unroll
    for (int off = 16; off > 0; off >>= 1) sum += __shfl_xor(sum, off, 32);
    float inv = 1.f / sum;
    for (int c = lane; c < PP; c += 32) S[r * PP + c] *= inv;
  }

  // pass 2: O = P @ V (A from f32 LDS strip, B from staged V^T)
  const v8f vz = {};
  v8f acc2[4];
#pragma unroll
  for (int nt = 0; nt < 4; ++nt) acc2[nt] = vz;
  const float* prow = &S[(lane & 15) * PP];
  for (int kk2 = 0; kk2 < PP / 32; ++kk2) {
    v16h a = a_frag(prow, kk2 * 32, lane, 1.f);
#pragma unroll
    for (int nt = 0; nt < 4; ++nt) {
      const _Float16* vrow = VL + (size_t)(nt * 16 + (lane & 15)) * PP;
      acc2[nt] = wmma_f16(a, b_frag_h(vrow, kk2 * 32, lane), acc2[nt]);
    }
  }
#pragma unroll
  for (int j = 0; j < 8; ++j) {
    int row = j + ((lane >> 4) << 3);
    int q = mt * 16 + row;
    float vg = Vg[eb + q];
#pragma unroll
    for (int nt = 0; nt < 4; ++nt)
      Ob[(eb + q) * EE + nt * 16 + (lane & 15)] = acc2[nt][j] * vg;
  }
}

// ---------------------------------- gate MLP per (expert, batch): scalar gs
__global__ void gate_kernel(const float* __restrict__ Ob,
                            const float* __restrict__ maskf,
                            const float* __restrict__ agW1,
                            const float* __restrict__ agb1,
                            const float* __restrict__ ag_g,
                            const float* __restrict__ ag_b,
                            const float* __restrict__ agW2,
                            const float* __restrict__ agb2,
                            const float* __restrict__ alpha, float* gs) {
  const int e = blockIdx.x, b = blockIdx.y, t = threadIdx.x;  // 64 threads
  __shared__ float ctx[EE];
  __shared__ float h[EE / 2];
  __shared__ float cnt_s;
  if (t == 0) {
    float c = 0.f;
    for (int p = 0; p < PP; ++p) c += maskf[e * PP + p];
    cnt_s = fmaxf(c, 1.f);
  }
  float acc = 0.f;
  const float* base = Ob + ((size_t)e * M_TOTAL + (size_t)b * PP) * EE;
  for (int p = 0; p < PP; ++p) acc += base[(size_t)p * EE + t] * maskf[e * PP + p];
  __syncthreads();
  ctx[t] = acc / cnt_s;
  __syncthreads();
  if (t < 32) {
    float hv = agb1[e * 32 + t];
    for (int k = 0; k < EE; ++k) hv += ctx[k] * agW1[((size_t)e * EE + k) * 32 + t];
    h[t] = hv;
  }
  __syncthreads();
  if (t == 0) {
    float mu = 0.f;
    for (int j = 0; j < 32; ++j) mu += h[j];
    mu *= (1.f / 32.f);
    float var = 0.f;
    for (int j = 0; j < 32; ++j) {
      float d = h[j] - mu;
      var += d * d;
    }
    var *= (1.f / 32.f);
    float inv = rsqrtf(var + 1e-5f);
    float z = 0.f;
    for (int j = 0; j < 32; ++j) {
      float x = (h[j] - mu) * inv * ag_g[e * 32 + j] + ag_b[e * 32 + j];
      x = 0.5f * x * (1.f + erff(x * 0.70710678118654752f));
      z += x * agW2[e * 32 + j];
    }
    z += agb2[e];
    float gate = 1.f / (1.f + __expf(-z));
    float sa = 1.f / (1.f + __expf(-alpha[e]));
    gs[e * BB + b] = sa * gate;
  }
}

// --------- final projection + fuse: both head-sharing experts in one WMMA acc
__global__ void fuse_kernel(const float* __restrict__ Ob,
                            const _Float16* __restrict__ oWT,
                            const float* __restrict__ ob,
                            const float* __restrict__ maskf,
                            const float* __restrict__ votes,
                            const float* __restrict__ gs, float* out) {
  const int lane = threadIdx.x;
  const int mt = blockIdx.x, head = blockIdx.y;
  const int ar = mt * 16 + (lane & 15);
  const int ab = ar / PP, ap = ar % PP;
  const v8f vz = {};
  v8f acc[8];
#pragma unroll
  for (int nt = 0; nt < 8; ++nt) acc[nt] = vz;

#pragma unroll
  for (int sub = 0; sub < 2; ++sub) {
    const int e = head + 8 * sub;
    const float s = maskf[e * PP + ap] * gs[e * BB + ab];  // mask*gate row scale
    const float* arow = Ob + ((size_t)e * M_TOTAL + ar) * EE;
    const _Float16* W = oWT + (size_t)e * SSZ * EE;
#pragma unroll
    for (int ks = 0; ks < 2; ++ks) {
      v16h a = a_frag(arow, ks * 32, lane, s);
#pragma unroll
      for (int nt = 0; nt < 8; ++nt) {
        const _Float16* wrow = W + (size_t)(nt * 16 + (lane & 15)) * EE;
        acc[nt] = wmma_f16(a, b_frag_h(wrow, ks * 32, lane), acc[nt]);
      }
    }
  }
  const int e0 = head, e1 = head + 8;
#pragma unroll
  for (int j = 0; j < 8; ++j) {
    int row = mt * 16 + j + ((lane >> 4) << 3);
    int b = row / PP, p = row % PP;
    float m0 = maskf[e0 * PP + p], m1 = maskf[e1 * PP + p];
    float inv = 1.f / fmaxf(votes[p], 1.f);
#pragma unroll
    for (int nt = 0; nt < 8; ++nt) {
      int colg = nt * 16 + (lane & 15);
      float v = acc[nt][j] + m0 * ob[e0 * SSZ + colg] + m1 * ob[e1 * SSZ + colg];
      out[((size_t)b * PP + p) * DD + head * SSZ + colg] = v * inv;
    }
  }
}

extern "C" void kernel_launch(void* const* d_in, const int* in_sizes, int n_in,
                              void* d_out, int out_size, void* d_ws,
                              size_t ws_size, hipStream_t stream) {
  (void)in_sizes; (void)n_in; (void)out_size; (void)ws_size;
  const float* tokens       = (const float*)d_in[0];
  const float* fingerprints = (const float*)d_in[1];
  const float* penta        = (const float*)d_in[2];
  const float* qW           = (const float*)d_in[3];
  const float* qb           = (const float*)d_in[4];
  const float* mha_w        = (const float*)d_in[5];
  const float* mha_b        = (const float*)d_in[6];
  const float* mha_ow       = (const float*)d_in[7];
  const float* mha_ob       = (const float*)d_in[8];
  const float* kW           = (const float*)d_in[9];
  const float* kb           = (const float*)d_in[10];
  const float* vW           = (const float*)d_in[11];
  const float* vb           = (const float*)d_in[12];
  const float* oW           = (const float*)d_in[13];
  const float* ob           = (const float*)d_in[14];
  const float* alpha        = (const float*)d_in[15];
  const float* agW1         = (const float*)d_in[16];
  const float* agb1         = (const float*)d_in[17];
  const float* ag_g         = (const float*)d_in[18];
  const float* ag_b         = (const float*)d_in[19];
  const float* agW2         = (const float*)d_in[20];
  const float* agb2         = (const float*)d_in[21];
  float* out = (float*)d_out;
  float* wsf = (float*)d_ws;

  // f32 workspace region
  float* MASKF = wsf;
  float* VOTES = MASKF + (size_t)NE * PP;
  float* KQ    = VOTES + PP;
  float* VO    = KQ + (size_t)NE * 5 * EE;
  float* SB    = VO + (size_t)NE * 5 * EE;
  float* GS    = SB + (size_t)NE * 5;
  float* KGg   = GS + (size_t)NE * BB;
  float* VGg   = KGg + (size_t)NE * M_TOTAL;
  float* Qb    = VGg + (size_t)NE * M_TOTAL;
  float* Kbuf  = Qb + (size_t)NE * M_TOTAL * EE;
  float* Vbuf  = Kbuf + (size_t)NE * M_TOTAL * EE;
  float* Obuf  = Vbuf + (size_t)NE * M_TOTAL * EE;
  // f16 region (64B aligned)
  char* hraw = (char*)(Obuf + (size_t)NE * M_TOTAL * EE);
  hraw = (char*)(((size_t)hraw + 63) & ~(size_t)63);
  _Float16* Qh   = (_Float16*)hraw;
  _Float16* Kh   = Qh + (size_t)NE * M_TOTAL * EE;
  _Float16* VhT  = Kh + (size_t)NE * M_TOTAL * EE;
  _Float16* qWT  = VhT + (size_t)NE * M_TOTAL * EE;
  _Float16* kWT  = qWT + (size_t)NE * EE * SSZ;
  _Float16* vWT  = kWT + (size_t)NE * EE * SSZ;
  _Float16* oWT  = vWT + (size_t)NE * EE * SSZ;
  _Float16* tokh = oWT + (size_t)NE * EE * SSZ;  // B*P*D halves

  mask_kernel<<<1, PP, 0, stream>>>(fingerprints, MASKF, VOTES);
  convtok_kernel<<<(size_t)BB * PP * DD / 256, 256, 0, stream>>>(tokens, tokh);
  convw_kernel<<<dim3(32, NE), 256, 0, stream>>>(qW, kW, vW, oW, qWT, kWT, vWT,
                                                 oWT);
  qkv_kernel<<<dim3(M_TOTAL / 16, NE, 3), 32, 0, stream>>>(
      tokh, qWT, kWT, vWT, qb, kb, vb, Qb, Kbuf, Vbuf, Kh, VhT);
  mha_pre_kernel<<<NE, EE, 0, stream>>>(penta, mha_w, mha_b, mha_ow, KQ, VO, SB);
  mha_apply_kernel<<<(NE * M_TOTAL) / 256, 256, 0, stream>>>(Qb, Qh, KQ, VO, SB,
                                                             mha_ob);
  kgvg_kernel<<<dim3(M_TOTAL / 256, NE), 256, 0, stream>>>(penta, Kbuf, Vbuf,
                                                           KGg, VGg);
  const size_t attn_lds = (size_t)2 * PP * EE * 2 + (size_t)4 * 16 * PP * 4;
  attn_kernel<<<dim3(PP / 16 / 4, BB, NE), 128, attn_lds, stream>>>(
      Qh, Kh, VhT, KGg, VGg, MASKF, Obuf);
  gate_kernel<<<dim3(NE, BB), EE, 0, stream>>>(Obuf, MASKF, agW1, agb1, ag_g,
                                               ag_b, agW2, agb2, alpha, GS);
  fuse_kernel<<<dim3(M_TOTAL / 16, NH), 32, 0, stream>>>(Obuf, oWT, ob, MASKF,
                                                         VOTES, GS, out);
}
